// alighnment_score_70686571758009
// MI455X (gfx1250) — matrix operand order repaired and checked by reading, not compiled, not run on hardware
//
#include <hip/hip_runtime.h>
#include <hip/hip_bf16.h>
#include <stdint.h>
#include <stddef.h>

#define DD 768
#define BB 8
#define SS 2048
#define RSQRT_D 0.03608439182435161f   // 1/sqrt(768)

typedef __attribute__((ext_vector_type(8)))  __bf16 bf16x8;
typedef __attribute__((ext_vector_type(16))) __bf16 bf16x16;
typedef __attribute__((ext_vector_type(8)))  float  f32x8;
typedef __attribute__((ext_vector_type(4)))  unsigned int u32x4;
typedef __attribute__((ext_vector_type(8)))  int i32x8;
typedef __attribute__((ext_vector_type(4)))  int i32x4;

union ABArg { bf16x16 v; bf16x8 h[2]; };

#if defined(__has_builtin)
#if __has_builtin(__builtin_amdgcn_tensor_load_to_lds) && __has_builtin(__builtin_amdgcn_s_wait_tensorcnt)
#define HAVE_TDM 1
#endif
#endif

__device__ __forceinline__ unsigned short f2bf(float f) {
  union { float f; unsigned int u; } v; v.f = f;
  unsigned int r = 0x7FFFu + ((v.u >> 16) & 1u);
  return (unsigned short)((v.u + r) >> 16);
}

// Padded LDS address: insert 16B pad every (1<<shift) bytes (matches TDM pad_amount=4dw)
__device__ __forceinline__ unsigned lds_pad(unsigned l, unsigned shift) {
  return l + ((l >> shift) << 4);
}

#ifdef HAVE_TDM
// TDM: load a rows x cols (bf16) contiguous tile (stride==cols) into LDS with padding.
// padint_code: 6 => pad every 128 dwords (512B); 7 => every 256 dwords (1024B).
// pad amount fixed at 4 dwords (16B) => padamt_code 3.
__device__ __forceinline__ void tdm_load_tile(unsigned lds_off, const void* gptr,
                                              unsigned cols, unsigned rows,
                                              unsigned padint_code) {
  unsigned long long ga = (unsigned long long)gptr;
  u32x4 g0;
  g0[0] = 1u;                                       // count=1 (user descriptor)
  g0[1] = lds_off;                                  // lds_addr (bytes)
  g0[2] = (unsigned)ga;                             // global_addr[31:0]
  g0[3] = (unsigned)((ga >> 32) & 0x1FFFFFFull) | (2u << 30);  // addr[56:32] | type=2
  i32x8 g1;
  g1[0] = (int)((1u << 16) | (1u << 20) | (padint_code << 22) | (3u << 25));
  // data_size=1(2B)  pad_enable     pad_interval            pad_amount=4dw
  g1[1] = (int)(cols << 16);      // tensor_dim0[15:0] in [31:16]
  g1[2] = (int)(rows << 16);      // tensor_dim0 hi=0 ; tensor_dim1[15:0]=rows
  g1[3] = (int)(cols << 16);      // tensor_dim1 hi=0 ; tile_dim0=cols
  g1[4] = (int)rows;              // tile_dim1=rows ; tile_dim2=0
  g1[5] = (int)cols;              // tensor_dim0_stride[31:0] = cols
  g1[6] = 0; g1[7] = 0;
  i32x4 g2 = {0, 0, 0, 0}, g3 = {0, 0, 0, 0};
  i32x8 g4 = {0, 0, 0, 0, 0, 0, 0, 0};
  // 6-arg toolchain variant (clang-23 / therock-10.0 headers)
  __builtin_amdgcn_tensor_load_to_lds(g0, g1, g2, g3, g4, 0);
}
#endif

// Fallback / reference copy into the identical padded LDS layout.
__device__ __forceinline__ void coop_load_tile(char* lds, const unsigned short* g,
                                               unsigned rows, unsigned cols,
                                               unsigned shift, unsigned nthr, unsigned tid) {
  unsigned chunks = rows * cols / 8;    // 16B chunks, tile is contiguous in global
  for (unsigned c = tid; c < chunks; c += nthr) {
    unsigned l = c * 16u;
    *(bf16x8*)(lds + lds_pad(l, shift)) = *(const bf16x8*)(g + c * 8);
  }
}

// Stage tile (issued by wave 0 via TDM if available, else cooperatively), then barrier.
__device__ __forceinline__ void stage_tile(char* lds, unsigned lds_off, const unsigned short* g,
                                           unsigned rows, unsigned cols, unsigned shift,
                                           unsigned padint_code, unsigned nthr) {
#ifdef HAVE_TDM
  if (threadIdx.x < 32) {
    tdm_load_tile(lds_off, (const void*)g, cols, rows, padint_code);
    __builtin_amdgcn_s_wait_tensorcnt(0);
  }
#else
  coop_load_tile(lds, g, rows, cols, shift, nthr, threadIdx.x);
#endif
  __syncthreads();
}

// ---------------- 1) LayerNorm -> xn (bf16) ----------------
__global__ void __launch_bounds__(256) ln_kernel(const float* __restrict__ x,
                                                 const float* __restrict__ g,
                                                 const float* __restrict__ bta,
                                                 unsigned short* __restrict__ xn) {
  int row = blockIdx.x;
  const float* xr = x + (size_t)row * DD;
  int t = threadIdx.x;
  float v0 = xr[t], v1 = xr[t + 256], v2 = xr[t + 512];
  __shared__ float red[256], red2[256];
  red[t]  = v0 + v1 + v2;
  red2[t] = v0 * v0 + v1 * v1 + v2 * v2;
  __syncthreads();
  for (int off = 128; off > 0; off >>= 1) {
    if (t < off) { red[t] += red[t + off]; red2[t] += red2[t + off]; }
    __syncthreads();
  }
  float mu  = red[0] * (1.0f / DD);
  float var = red2[0] * (1.0f / DD) - mu * mu;
  float inv = rsqrtf(var + 1e-5f);
  unsigned short* o = xn + (size_t)row * DD;
  o[t]       = f2bf((v0 - mu) * inv * g[t]       + bta[t]);
  o[t + 256] = f2bf((v1 - mu) * inv * g[t + 256] + bta[t + 256]);
  o[t + 512] = f2bf((v2 - mu) * inv * g[t + 512] + bta[t + 512]);
}

// ---------------- 2) weights f32 -> bf16 ----------------
__global__ void wcvt_kernel(const float* __restrict__ wq, const float* __restrict__ wk,
                            unsigned short* __restrict__ wqb, unsigned short* __restrict__ wkb) {
  int i = blockIdx.x * blockDim.x + threadIdx.x;
  if (i < DD * DD) { wqb[i] = f2bf(wq[i]); wkb[i] = f2bf(wk[i]); }
}

// ---------------- 3) LDS-tiled transpose x -> xT (bf16), xT[b][d][s] ----------------
__global__ void xt_kernel(const float* __restrict__ x, unsigned short* __restrict__ xT) {
  __shared__ unsigned short tile[32][33];
  int bz = blockIdx.z, bx = blockIdx.x, by = blockIdx.y;
  int tx = threadIdx.x, ty = threadIdx.y;                  // (32,8)
  const float* xb = x + (size_t)bz * SS * DD;
#pragma unroll
  for (int i = 0; i < 4; i++) {
    int s = bx * 32 + ty + i * 8;
    int d = by * 32 + tx;
    tile[ty + i * 8][tx] = f2bf(xb[(size_t)s * DD + d]);
  }
  __syncthreads();
  unsigned short* xtb = xT + (size_t)bz * DD * SS;
#pragma unroll
  for (int i = 0; i < 4; i++) {
    int d = by * 32 + ty + i * 8;
    int s = bx * 32 + tx;
    xtb[(size_t)d * SS + s] = tile[tx][ty + i * 8];
  }
}

// ---------------- 4) Q/K projection; A tile (xn) staged via TDM ----------------
// 16x768 bf16 tile, padded rows: 1536B + 3*16B = 1584B -> 25344B LDS.
__global__ void __launch_bounds__(256) qkproj_kernel(
    const unsigned short* __restrict__ xn,
    const unsigned short* __restrict__ wqb, const unsigned short* __restrict__ wkb,
    const float* __restrict__ bq, const float* __restrict__ bk,
    unsigned short* __restrict__ qo, unsigned short* __restrict__ ko) {
  __shared__ __align__(16) char lds_a[16 * (DD * 2 + 48)];
  int mt = blockIdx.x / 6, ng = blockIdx.x % 6;
  int warp = threadIdx.x >> 5;
  int nt = ng * 8 + warp;                                  // 0..47
  int lane = threadIdx.x & 31, lo = lane & 15, hi = lane >> 4;

  stage_tile(lds_a, (unsigned)(size_t)(&lds_a[0]),
             xn + (size_t)mt * 16 * DD, 16, DD, 9, 6, 256);

  const unsigned short* Bq = wqb + (size_t)(nt * 16 + lo) * DD;
  const unsigned short* Bk = wkb + (size_t)(nt * 16 + lo) * DD;
  f32x8 accq = {}, acck = {};
  for (int k = 0; k < DD; k += 32) {
    ABArg a, b1, b2;
    unsigned l0 = (unsigned)lo * (DD * 2) + (unsigned)(k + hi * 16) * 2u;
    unsigned l1 = (unsigned)lo * (DD * 2) + (unsigned)(k + hi * 16 + 8) * 2u;
    a.h[0]  = *(const bf16x8*)(lds_a + lds_pad(l0, 9));
    a.h[1]  = *(const bf16x8*)(lds_a + lds_pad(l1, 9));
    b1.h[0] = *(const bf16x8*)(Bq + k + hi * 16);
    b1.h[1] = *(const bf16x8*)(Bq + k + hi * 16 + 8);
    b2.h[0] = *(const bf16x8*)(Bk + k + hi * 16);
    b2.h[1] = *(const bf16x8*)(Bk + k + hi * 16 + 8);
    accq = __builtin_amdgcn_wmma_f32_16x16x32_bf16(false, a.v, false, b1.v, (short)0, accq, false, false);
    acck = __builtin_amdgcn_wmma_f32_16x16x32_bf16(false, a.v, false, b2.v, (short)0, acck, false, false);
  }
  int n = nt * 16 + lo;
  float biasq = bq[n], biask = bk[n];
#pragma unroll
  for (int r = 0; r < 8; r++) {
    int m = mt * 16 + hi * 8 + r;
    qo[(size_t)m * DD + n] = f2bf(accq[r] + biasq);
    ko[(size_t)m * DD + n] = f2bf(acck[r] + biask);
  }
}

// ---------------- 5) scores = Q @ K^T * rsqrt(D); Q tile staged via TDM ----------------
// Each wave: 16x64 (4 accumulators). Block: 8 waves -> 512 score columns.
__global__ void __launch_bounds__(256) scores_kernel(
    const unsigned short* __restrict__ q, const unsigned short* __restrict__ kmat,
    float* __restrict__ sc) {
  __shared__ __align__(16) char lds_a[16 * (DD * 2 + 48)];
  int bid = blockIdx.x;                     // 8 * 128 * 4
  int b  = bid >> 9;
  int r  = bid & 511;
  int mt = r >> 2, ng = r & 3;
  int warp = threadIdx.x >> 5;
  int lane = threadIdx.x & 31, lo = lane & 15, hi = lane >> 4;
  int n0 = ng * 512 + warp * 64;

  stage_tile(lds_a, (unsigned)(size_t)(&lds_a[0]),
             q + ((size_t)b * SS + mt * 16) * DD, 16, DD, 9, 6, 256);

  const unsigned short* B0 = kmat + ((size_t)b * SS + n0 +  0 + lo) * DD;
  const unsigned short* B1 = kmat + ((size_t)b * SS + n0 + 16 + lo) * DD;
  const unsigned short* B2 = kmat + ((size_t)b * SS + n0 + 32 + lo) * DD;
  const unsigned short* B3 = kmat + ((size_t)b * SS + n0 + 48 + lo) * DD;
  f32x8 acc0 = {}, acc1 = {}, acc2 = {}, acc3 = {};
  for (int k = 0; k < DD; k += 32) {
    ABArg a, b0, b1, b2, b3;
    unsigned l0 = (unsigned)lo * (DD * 2) + (unsigned)(k + hi * 16) * 2u;
    unsigned l1 = (unsigned)lo * (DD * 2) + (unsigned)(k + hi * 16 + 8) * 2u;
    a.h[0] = *(const bf16x8*)(lds_a + lds_pad(l0, 9));
    a.h[1] = *(const bf16x8*)(lds_a + lds_pad(l1, 9));
    b0.h[0] = *(const bf16x8*)(B0 + k + hi * 16); b0.h[1] = *(const bf16x8*)(B0 + k + hi * 16 + 8);
    b1.h[0] = *(const bf16x8*)(B1 + k + hi * 16); b1.h[1] = *(const bf16x8*)(B1 + k + hi * 16 + 8);
    b2.h[0] = *(const bf16x8*)(B2 + k + hi * 16); b2.h[1] = *(const bf16x8*)(B2 + k + hi * 16 + 8);
    b3.h[0] = *(const bf16x8*)(B3 + k + hi * 16); b3.h[1] = *(const bf16x8*)(B3 + k + hi * 16 + 8);
    acc0 = __builtin_amdgcn_wmma_f32_16x16x32_bf16(false, a.v, false, b0.v, (short)0, acc0, false, false);
    acc1 = __builtin_amdgcn_wmma_f32_16x16x32_bf16(false, a.v, false, b1.v, (short)0, acc1, false, false);
    acc2 = __builtin_amdgcn_wmma_f32_16x16x32_bf16(false, a.v, false, b2.v, (short)0, acc2, false, false);
    acc3 = __builtin_amdgcn_wmma_f32_16x16x32_bf16(false, a.v, false, b3.v, (short)0, acc3, false, false);
  }
  float* sb = sc + (size_t)b * SS * SS;
#pragma unroll
  for (int j = 0; j < 4; j++) {
    f32x8 acc = (j == 0) ? acc0 : (j == 1) ? acc1 : (j == 2) ? acc2 : acc3;
    int n = n0 + j * 16 + lo;
#pragma unroll
    for (int rr = 0; rr < 8; rr++) {
      int m = mt * 16 + hi * 8 + rr;
      sb[(size_t)m * SS + n] = acc[rr] * RSQRT_D;
    }
  }
}

// ---------------- 6) row softmax -> P (bf16) ----------------
__global__ void __launch_bounds__(256) softmax_kernel(const float* __restrict__ sc,
                                                      unsigned short* __restrict__ p) {
  int row = blockIdx.x;
  const float* sr = sc + (size_t)row * SS;
  unsigned short* pr = p + (size_t)row * SS;
  int t = threadIdx.x;
  float v[8]; float mx = -3.0e38f;
#pragma unroll
  for (int i = 0; i < 8; i++) { v[i] = sr[t + i * 256]; mx = fmaxf(mx, v[i]); }
  __shared__ float red[256];
  red[t] = mx; __syncthreads();
  for (int off = 128; off > 0; off >>= 1) {
    if (t < off) red[t] = fmaxf(red[t], red[t + off]);
    __syncthreads();
  }
  mx = red[0]; __syncthreads();
  float sum = 0.f;
#pragma unroll
  for (int i = 0; i < 8; i++) { v[i] = __expf(v[i] - mx); sum += v[i]; }
  red[t] = sum; __syncthreads();
  for (int off = 128; off > 0; off >>= 1) {
    if (t < off) red[t] += red[t + off];
    __syncthreads();
  }
  float inv = 1.0f / red[0];
#pragma unroll
  for (int i = 0; i < 8; i++) pr[t + i * 256] = f2bf(v[i] * inv);
}

// ---------------- 7) out = P @ x + x; P tile staged via TDM, shared by 12 waves ----------------
// P tile 16x2048 bf16, padded rows: 4096B + 4*16B = 4160B -> 66560B LDS.
__global__ void __launch_bounds__(384) pv_kernel(
    const unsigned short* __restrict__ p, const unsigned short* __restrict__ xT,
    const float* __restrict__ x, float* __restrict__ out) {
  __shared__ __align__(16) char lds_a[16 * (SS * 2 + 64)];
  int b  = blockIdx.x >> 7;
  int mt = blockIdx.x & 127;
  int warp = threadIdx.x >> 5;              // 0..11 => n0 = warp*64
  int lane = threadIdx.x & 31, lo = lane & 15, hi = lane >> 4;
  int n0 = warp * 64;

  stage_tile(lds_a, (unsigned)(size_t)(&lds_a[0]),
             p + ((size_t)b * SS + mt * 16) * SS, 16, SS, 10, 7, 384);

  const unsigned short* Bb = xT + (size_t)b * DD * SS;
  const unsigned short* B0 = Bb + (size_t)(n0 +  0 + lo) * SS;
  const unsigned short* B1 = Bb + (size_t)(n0 + 16 + lo) * SS;
  const unsigned short* B2 = Bb + (size_t)(n0 + 32 + lo) * SS;
  const unsigned short* B3 = Bb + (size_t)(n0 + 48 + lo) * SS;
  f32x8 acc0 = {}, acc1 = {}, acc2 = {}, acc3 = {};
  for (int k = 0; k < SS; k += 32) {
    ABArg a, b0, b1, b2, b3;
    unsigned l0 = (unsigned)lo * (SS * 2) + (unsigned)(k + hi * 16) * 2u;
    unsigned l1 = (unsigned)lo * (SS * 2) + (unsigned)(k + hi * 16 + 8) * 2u;
    a.h[0] = *(const bf16x8*)(lds_a + lds_pad(l0, 10));
    a.h[1] = *(const bf16x8*)(lds_a + lds_pad(l1, 10));
    b0.h[0] = *(const bf16x8*)(B0 + k + hi * 16); b0.h[1] = *(const bf16x8*)(B0 + k + hi * 16 + 8);
    b1.h[0] = *(const bf16x8*)(B1 + k + hi * 16); b1.h[1] = *(const bf16x8*)(B1 + k + hi * 16 + 8);
    b2.h[0] = *(const bf16x8*)(B2 + k + hi * 16); b2.h[1] = *(const bf16x8*)(B2 + k + hi * 16 + 8);
    b3.h[0] = *(const bf16x8*)(B3 + k + hi * 16); b3.h[1] = *(const bf16x8*)(B3 + k + hi * 16 + 8);
    acc0 = __builtin_amdgcn_wmma_f32_16x16x32_bf16(false, a.v, false, b0.v, (short)0, acc0, false, false);
    acc1 = __builtin_amdgcn_wmma_f32_16x16x32_bf16(false, a.v, false, b1.v, (short)0, acc1, false, false);
    acc2 = __builtin_amdgcn_wmma_f32_16x16x32_bf16(false, a.v, false, b2.v, (short)0, acc2, false, false);
    acc3 = __builtin_amdgcn_wmma_f32_16x16x32_bf16(false, a.v, false, b3.v, (short)0, acc3, false, false);
  }
  const float* xb = x + (size_t)b * SS * DD;
  float* ob = out + (size_t)b * SS * DD;
#pragma unroll
  for (int j = 0; j < 4; j++) {
    f32x8 acc = (j == 0) ? acc0 : (j == 1) ? acc1 : (j == 2) ? acc2 : acc3;
    int n = n0 + j * 16 + lo;
#pragma unroll
    for (int rr = 0; rr < 8; rr++) {
      int m = mt * 16 + hi * 8 + rr;
      ob[(size_t)m * DD + n] = acc[rr] + xb[(size_t)m * DD + n];
    }
  }
}

extern "C" void kernel_launch(void* const* d_in, const int* in_sizes, int n_in,
                              void* d_out, int out_size, void* d_ws, size_t ws_size,
                              hipStream_t stream) {
  const float* x    = (const float*)d_in[0];
  const float* ln_g = (const float*)d_in[1];
  const float* ln_b = (const float*)d_in[2];
  const float* Wq   = (const float*)d_in[3];
  const float* bq   = (const float*)d_in[4];
  const float* Wk   = (const float*)d_in[5];
  const float* bk   = (const float*)d_in[6];
  float* out = (float*)d_out;
  char* ws = (char*)d_ws;

  const size_t n_bsd = (size_t)BB * SS * DD;
  const size_t off_xn = 0;
  const size_t off_q  = off_xn + n_bsd * 2;
  const size_t off_k  = off_q  + n_bsd * 2;
  const size_t off_xT = off_k  + n_bsd * 2;
  const size_t off_wq = off_xT + n_bsd * 2;
  const size_t off_wk = off_wq + (size_t)DD * DD * 2;
  const size_t off_sc = off_wk + (size_t)DD * DD * 2;
  const size_t off_p  = off_sc + (size_t)BB * SS * SS * 4;

  unsigned short* xn  = (unsigned short*)(ws + off_xn);
  unsigned short* qb  = (unsigned short*)(ws + off_q);
  unsigned short* kb  = (unsigned short*)(ws + off_k);
  unsigned short* xT  = (unsigned short*)(ws + off_xT);
  unsigned short* wqb = (unsigned short*)(ws + off_wq);
  unsigned short* wkb = (unsigned short*)(ws + off_wk);
  float*          sc  = (float*)(ws + off_sc);
  unsigned short* pm  = (unsigned short*)(ws + off_p);

  ln_kernel<<<BB * SS, 256, 0, stream>>>(x, ln_g, ln_b, xn);
  wcvt_kernel<<<(DD * DD + 255) / 256, 256, 0, stream>>>(Wq, Wk, wqb, wkb);
  xt_kernel<<<dim3(SS / 32, DD / 32, BB), dim3(32, 8), 0, stream>>>(x, xT);
  qkproj_kernel<<<1024 * 6, 256, 0, stream>>>(xn, wqb, wkb, bq, bk, qb, kb);
  scores_kernel<<<BB * 128 * 4, 256, 0, stream>>>(qb, kb, sc);
  softmax_kernel<<<BB * SS, 256, 0, stream>>>(sc, pm);
  pv_kernel<<<BB * 128, 384, 0, stream>>>(pm, xT, x, out);
}
